// CausalSelfAttention_83227876262054
// MI455X (gfx1250) — matrix operand order repaired and checked by reading, not compiled, not run on hardware
//
#include <hip/hip_runtime.h>

typedef __attribute__((ext_vector_type(16))) __bf16 v16bf;
typedef __attribute__((ext_vector_type(8)))  __bf16 v8bf;
typedef __attribute__((ext_vector_type(8)))  float  v8f;
typedef __attribute__((ext_vector_type(4)))  int    v4i;

#define CDIM 1024
#define TSEQ 2048

// Block tile: 256(M) x 128(N), 8 waves as 4x2 of 64x64 wave tiles.
#define A_ROWS 256
#define B_ROWS 128
#define LDSTRIDE 40                      // 32 data + 8 pad elems: conflict-free b128 row reads
#define PANEL ((A_ROWS + B_ROWS) * LDSTRIDE)   // 15360 elems per buffer; x2 buffers = 60 KB LDS

#define WMMA_BF16(a, b, c) __builtin_amdgcn_wmma_f32_16x16x32_bf16(false, (a), false, (b), (short)0, (c), false, false)

// ---- 16-byte global->LDS copy: async (ASYNCcnt, no VGPR round-trip) when available ----
#if __has_builtin(__builtin_amdgcn_global_load_async_to_lds_b128)
#define HAVE_ASYNC_LDS 1
typedef __attribute__((address_space(1))) v4i gv4i;
typedef __attribute__((address_space(3))) v4i lv4i;
#else
#define HAVE_ASYNC_LDS 0
#endif

static __device__ __forceinline__ void cp16(__bf16* l, const __bf16* g) {
#if HAVE_ASYNC_LDS
    __builtin_amdgcn_global_load_async_to_lds_b128((gv4i*)g, (lv4i*)l, 0, 0);
#else
    *(v8bf*)l = *(const v8bf*)g;
#endif
}
static __device__ __forceinline__ void async_fence() {
#if HAVE_ASYNC_LDS
#if __has_builtin(__builtin_amdgcn_s_wait_asynccnt)
    __builtin_amdgcn_s_wait_asynccnt(0);
#else
    asm volatile("s_wait_asynccnt 0x0" ::: "memory");
#endif
#endif
}

// ---- LDS fragment loaders (wave32, v_wmma_f32_16x16x32_bf16 layouts) ----
// A fragment (16x32, MxK): lane<16 -> M=lane,    K = {0..7, 16..23}
//                          lane>=16 -> M=lane-16, K = {8..15, 24..31}
static __device__ __forceinline__ v16bf lds_a(const __bf16* smA, int rbase, int lane) {
    const __bf16* r = smA + (size_t)(rbase + (lane & 15)) * LDSTRIDE + ((lane >> 4) * 8);
    v8bf lo = *(const v8bf*)(r);
    v8bf hi = *(const v8bf*)(r + 16);
    v16bf a;
#pragma unroll
    for (int i = 0; i < 8; ++i) { a[i] = lo[i]; a[i + 8] = hi[i]; }
    return a;
}
// B fragment (32x16, KxN): n = lane&15 row of panel, lane half picks K {0..7,16..23} / {8..15,24..31}.
static __device__ __forceinline__ v16bf lds_b(const __bf16* smB, int rbase, int lane) {
    const __bf16* r = smB + (size_t)(rbase + (lane & 15)) * LDSTRIDE + ((lane >> 4) * 8);
    v8bf lo = *(const v8bf*)(r);
    v8bf hi = *(const v8bf*)(r + 16);
    v16bf b;
#pragma unroll
    for (int i = 0; i < 8; ++i) { b[i] = lo[i]; b[i + 8] = hi[i]; }
    return b;
}

// ---- Cooperative staging: A panel 256x32 (1 row/thread), B panel 128x32 (1/2 row/thread) ----
static __device__ __forceinline__ void stage(__bf16* sm,
                                             const __bf16* Ag, int sA,
                                             const __bf16* Bg, int sB,
                                             int k0, int tid) {
    {
        const __bf16* g = Ag + (size_t)tid * sA + k0;
        __bf16* l = sm + tid * LDSTRIDE;
#pragma unroll
        for (int t = 0; t < 4; ++t)
            cp16(l + t * 8, g + t * 8);
    }
    {
        int row = tid >> 1, part = (tid & 1) * 16;
        const __bf16* g = Bg + (size_t)row * sB + k0 + part;
        __bf16* l = sm + A_ROWS * LDSTRIDE + row * LDSTRIDE + part;
        cp16(l,     g);
        cp16(l + 8, g + 8);
    }
}

// ---- Double-buffered 256x128 block GEMM; each wave accumulates a 64x64 tile.
//      klimit: wave-private compute cutoff (uniform branch; EXEC stays all-ones). ----
static __device__ __forceinline__ void block_gemm(
    const __bf16* Ag, int sA, const __bf16* Bg, int sB,
    int kend, int klimit, __bf16* sm, int mi, int ni, int lane, int tid,
    v8f acc[4][4])
{
    stage(sm, Ag, sA, Bg, sB, 0, tid);
    async_fence();
    __syncthreads();
    int buf = 0;
    for (int k0 = 0; k0 < kend; k0 += 32) {
        if (k0 + 32 < kend) stage(sm + (buf ^ 1) * PANEL, Ag, sA, Bg, sB, k0 + 32, tid);
        if (k0 < klimit) {
            const __bf16* smA = sm + buf * PANEL;
            const __bf16* smB = smA + A_ROWS * LDSTRIDE;
            v16bf a[4], bm[4];
#pragma unroll
            for (int i = 0; i < 4; ++i) a[i]  = lds_a(smA, mi * 64 + i * 16, lane);
#pragma unroll
            for (int j = 0; j < 4; ++j) bm[j] = lds_b(smB, ni * 64 + j * 16, lane);
#pragma unroll
            for (int i = 0; i < 4; ++i)
#pragma unroll
                for (int j = 0; j < 4; ++j) acc[i][j] = WMMA_BF16(a[i], bm[j], acc[i][j]);
        }
        async_fence();
        __syncthreads();
        buf ^= 1;
    }
}

// ---- Kernel 0: f32 -> bf16 bulk convert ----
__global__ __launch_bounds__(256) void k_cvt(const float* __restrict__ src, __bf16* __restrict__ dst, int n) {
    int i = (int)(blockIdx.x * 256 + threadIdx.x) * 8;
    if (i >= n) return;
    v8f v = *(const v8f*)(src + i);
    v8bf o;
#pragma unroll
    for (int k = 0; k < 8; ++k) o[k] = (__bf16)v[k];
    *(v8bf*)(dst + i) = o;
}

// ---- Kernel 1: Y = x W^T + b ; mode 0: row-major bf16, mode 1: transposed (B,C,T) ----
__global__ __launch_bounds__(256) void k_proj(
    const __bf16* __restrict__ xb, const __bf16* __restrict__ Wb,
    const float* __restrict__ bias, __bf16* __restrict__ dst, int transposed)
{
    __shared__ __bf16 sm[2 * PANEL];
    int tid = threadIdx.x, lane = tid & 31, wave = tid >> 5;
    int mi = wave >> 1, ni = wave & 1;
    int bm = (int)blockIdx.x >> 3, bn = (int)blockIdx.x & 7;   // 32 x 8 blocks
    int row0 = bm * 256, n0 = bn * 128;
    v8f acc[4][4] = {};
    block_gemm(xb + (size_t)row0 * CDIM, CDIM, Wb + (size_t)n0 * CDIM, CDIM,
               CDIM, CDIM, sm, mi, ni, lane, tid, acc);
    int nl = lane & 15, mbase = (lane >> 4) * 8;
    int wr0 = row0 + mi * 64, wn0 = n0 + ni * 64;
    if (!transposed) {
#pragma unroll
        for (int j = 0; j < 4; ++j) {
            float bj = bias[wn0 + j * 16 + nl];
#pragma unroll
            for (int i = 0; i < 4; ++i)
#pragma unroll
                for (int r = 0; r < 8; ++r)
                    dst[(size_t)(wr0 + i * 16 + mbase + r) * CDIM + wn0 + j * 16 + nl] =
                        (__bf16)(acc[i][j][r] + bj);
        }
    } else {
        int b  = wr0 / TSEQ;
        int t0 = wr0 % TSEQ;
        __bf16* vtb = dst + (size_t)b * CDIM * TSEQ;
#pragma unroll
        for (int j = 0; j < 4; ++j) {
            float bj = bias[wn0 + j * 16 + nl];
            __bf16* col = vtb + (size_t)(wn0 + j * 16 + nl) * TSEQ + t0;
#pragma unroll
            for (int i = 0; i < 4; ++i)
#pragma unroll
                for (int r = 0; r < 8; ++r)
                    col[i * 16 + mbase + r] = (__bf16)(acc[i][j][r] + bj);
        }
    }
}

// ---- Kernel 2: S = (Q K^T)/32, causal blocks only, f32 ----
__global__ __launch_bounds__(256) void k_scores(
    const __bf16* __restrict__ Qb, const __bf16* __restrict__ Kb, float* __restrict__ S)
{
    __shared__ __bf16 sm[2 * PANEL];
    int tid = threadIdx.x, lane = tid & 31, wave = tid >> 5;
    int mi = wave >> 1, ni = wave & 1;
    int b  = (int)blockIdx.x >> 7;                 // 8 x 16 blocks per batch
    int r2 = (int)blockIdx.x & 127;
    int tB = r2 >> 4, sB = r2 & 15;
    int t0 = tB * 256, s0 = sB * 128;
    if (s0 >= t0 + 256) return;                    // whole block above diagonal (uniform exit)
    v8f acc[4][4] = {};
    block_gemm(Qb + ((size_t)b * TSEQ + t0) * CDIM, CDIM,
               Kb + ((size_t)b * TSEQ + s0) * CDIM, CDIM,
               CDIM, CDIM, sm, mi, ni, lane, tid, acc);
    int nl = lane & 15, mbase = (lane >> 4) * 8;
    int wt0 = t0 + mi * 64, ws0 = s0 + ni * 64;
#pragma unroll
    for (int i = 0; i < 4; ++i)
#pragma unroll
        for (int r = 0; r < 8; ++r) {
            int t = wt0 + i * 16 + mbase + r;
            float* srow = S + ((size_t)b * TSEQ + t) * TSEQ;
#pragma unroll
            for (int j = 0; j < 4; ++j) {
                int s = ws0 + j * 16 + nl;
                float v = acc[i][j][r] * 0.03125f;          // 1/sqrt(1024)
                v = (s > t) ? -__builtin_inff() : v;
                srow[s] = v;
            }
        }
}

// ---- Kernel 3: row softmax; P written as bf16 in place over S, zero-padded to a 64-multiple ----
__global__ __launch_bounds__(256) void k_softmax(float* __restrict__ S)
{
    int wave = (int)((blockIdx.x * blockDim.x + threadIdx.x) >> 5);   // row 0..8191
    int lane = threadIdx.x & 31;
    int t = wave & (TSEQ - 1);
    float* p = S + (size_t)wave * TSEQ;
    int len = t + 1;
    float m = -__builtin_inff();
    for (int i = lane; i < len; i += 32) m = fmaxf(m, p[i]);
#pragma unroll
    for (int off = 16; off; off >>= 1) m = fmaxf(m, __shfl_xor(m, off, 32));
    float sum = 0.0f;
    for (int i = lane; i < len; i += 32) sum += __expf(p[i] - m);
#pragma unroll
    for (int off = 16; off; off >>= 1) sum += __shfl_xor(sum, off, 32);
    float inv = 1.0f / sum;
    int pad = ((t >> 6) + 1) << 6;                // multiple of 64: covers every PV k-range
    __bf16* q = (__bf16*)p;                       // bf16 region grows at half the f32 read rate: no clobber
    for (int i = lane; i < pad; i += 32) {
        float v = (i < len) ? __expf(p[i] - m) * inv : 0.0f;
        q[i] = (__bf16)v;
    }
}

// ---- Kernel 4: O = P @ V (V pre-transposed), causal K-range, wave-private klimit ----
__global__ __launch_bounds__(256) void k_pv(
    const __bf16* __restrict__ Pb, const __bf16* __restrict__ Vt, __bf16* __restrict__ Ob)
{
    __shared__ __bf16 sm[2 * PANEL];
    int tid = threadIdx.x, lane = tid & 31, wave = tid >> 5;
    int mi = wave >> 1, ni = wave & 1;
    int b  = (int)blockIdx.x >> 6;                 // 8 x 8 blocks per batch
    int r2 = (int)blockIdx.x & 63;
    int tB = r2 >> 3, dB = r2 & 7;
    int t0 = tB * 256, d0 = dB * 128;
    v8f acc[4][4] = {};
    int kend   = t0 + 256;                         // block causal K-range
    int klimit = t0 + mi * 64 + 64;                // this wave's rows only need cols < klimit
    block_gemm(Pb + ((size_t)b * TSEQ + t0) * (TSEQ * 2), TSEQ * 2,   // P aliases S rows (stride 4096)
               Vt + (size_t)b * CDIM * TSEQ + (size_t)d0 * TSEQ, TSEQ,
               kend, klimit, sm, mi, ni, lane, tid, acc);
    int nl = lane & 15, mbase = (lane >> 4) * 8;
    int wt0 = t0 + mi * 64, wd0 = d0 + ni * 64;
#pragma unroll
    for (int i = 0; i < 4; ++i)
#pragma unroll
        for (int r = 0; r < 8; ++r) {
            __bf16* orow = Ob + ((size_t)b * TSEQ + wt0 + i * 16 + mbase + r) * CDIM + wd0;
#pragma unroll
            for (int j = 0; j < 4; ++j)
                orow[j * 16 + nl] = (__bf16)acc[i][j][r];
        }
}

// ---- Kernel 5: out = O @ Wo^T + bo, f32 ----
__global__ __launch_bounds__(256) void k_oproj(
    const __bf16* __restrict__ Ob, const __bf16* __restrict__ Wob,
    const float* __restrict__ bo, float* __restrict__ out)
{
    __shared__ __bf16 sm[2 * PANEL];
    int tid = threadIdx.x, lane = tid & 31, wave = tid >> 5;
    int mi = wave >> 1, ni = wave & 1;
    int bm = (int)blockIdx.x >> 3, bn = (int)blockIdx.x & 7;
    int row0 = bm * 256, n0 = bn * 128;
    v8f acc[4][4] = {};
    block_gemm(Ob + (size_t)row0 * CDIM, CDIM, Wob + (size_t)n0 * CDIM, CDIM,
               CDIM, CDIM, sm, mi, ni, lane, tid, acc);
    int nl = lane & 15, mbase = (lane >> 4) * 8;
    int wr0 = row0 + mi * 64, wn0 = n0 + ni * 64;
#pragma unroll
    for (int j = 0; j < 4; ++j) {
        float bj = bo[wn0 + j * 16 + nl];
#pragma unroll
        for (int i = 0; i < 4; ++i)
#pragma unroll
            for (int r = 0; r < 8; ++r)
                out[(size_t)(wr0 + i * 16 + mbase + r) * CDIM + wn0 + j * 16 + nl] = acc[i][j][r] + bj;
    }
}

extern "C" void kernel_launch(void* const* d_in, const int* in_sizes, int n_in,
                              void* d_out, int out_size, void* d_ws, size_t ws_size,
                              hipStream_t stream) {
    (void)in_sizes; (void)n_in; (void)out_size; (void)ws_size;
    const float* x  = (const float*)d_in[0];
    const float* Wq = (const float*)d_in[1];
    const float* bq = (const float*)d_in[2];
    const float* Wk = (const float*)d_in[3];
    const float* bk = (const float*)d_in[4];
    const float* Wv = (const float*)d_in[5];
    const float* bv = (const float*)d_in[6];
    const float* Wo = (const float*)d_in[7];
    const float* bo = (const float*)d_in[8];
    float* out = (float*)d_out;

    // Workspace layout (~152 MB; hot set fits in the 192 MB L2):
    char* ws = (char*)d_ws;
    const size_t MB = (size_t)1 << 20;
    __bf16* xb  = (__bf16*)(ws);              // 16 MB  (8192 x 1024 bf16)
    __bf16* Wqb = (__bf16*)(ws + 16 * MB);    //  2 MB
    __bf16* Wkb = (__bf16*)(ws + 18 * MB);    //  2 MB
    __bf16* Wvb = (__bf16*)(ws + 20 * MB);    //  2 MB
    __bf16* Wob = (__bf16*)(ws + 22 * MB);    //  2 MB
    __bf16* Qb  = (__bf16*)(ws + 24 * MB);    // 16 MB
    __bf16* Kb  = (__bf16*)(ws + 40 * MB);    // 16 MB
    __bf16* Vt  = (__bf16*)(ws + 56 * MB);    // 16 MB  (B x C x T)
    float*  S   = (float*) (ws + 72 * MB);    // 64 MB  (8192 x 2048 f32; P aliases rows as bf16)
    __bf16* Ob  = (__bf16*)(ws + 136 * MB);   // 16 MB

    const int NX = 8192 * 1024, NW = 1024 * 1024;
    k_cvt<<<NX / (8 * 256), 256, 0, stream>>>(x,  xb,  NX);
    k_cvt<<<NW / (8 * 256), 256, 0, stream>>>(Wq, Wqb, NW);
    k_cvt<<<NW / (8 * 256), 256, 0, stream>>>(Wk, Wkb, NW);
    k_cvt<<<NW / (8 * 256), 256, 0, stream>>>(Wv, Wvb, NW);
    k_cvt<<<NW / (8 * 256), 256, 0, stream>>>(Wo, Wob, NW);

    k_proj   <<<256, 256, 0, stream>>>(xb, Wqb, bq, Qb, 0);
    k_proj   <<<256, 256, 0, stream>>>(xb, Wkb, bk, Kb, 0);
    k_proj   <<<256, 256, 0, stream>>>(xb, Wvb, bv, Vt, 1);
    k_scores <<<512, 256, 0, stream>>>(Qb, Kb, S);
    k_softmax<<<1024, 256, 0, stream>>>(S);
    k_pv     <<<256, 256, 0, stream>>>((const __bf16*)S, Vt, Ob);
    k_oproj  <<<256, 256, 0, stream>>>(Ob, Wob, bo, out);
}
